// DGCNNWithColor_26542897889434
// MI455X (gfx1250) — matrix operand order repaired and checked by reading, not compiled, not run on hardware
//
#include <hip/hip_runtime.h>
#include <hip/hip_bf16.h>

typedef __attribute__((ext_vector_type(16))) _Float16 v16h;
typedef __attribute__((ext_vector_type(8)))  float    v8f;

#define NPTS  4096
#define BATCH 4
#define KNN_K 20

// ---------------- WMMA fragment helpers (16x16x32 f16 -> f32) ----------------
// A-matrix 16x32 f16 layout (matches B): lanes 0-15 hold M=lane, K in
// k0..k0+15; lanes 16-31 hold M=lane-16, K in k0+16..k0+31. Two contiguous
// 8-half (16B) loads per lane.
__device__ __forceinline__ v16h load_a_frag(const _Float16* base, int stride,
                                            int row0, int k0, int lane) {
  int m  = lane & 15;
  int kk = k0 + ((lane & 16) ? 16 : 0);
  const _Float16* p0 = base + (size_t)(row0 + m) * stride + kk;
  const _Float16* p1 = p0 + 8;
  v16h a;
#pragma unroll
  for (int i = 0; i < 8; ++i) { a[i] = p0[i]; a[8 + i] = p1[i]; }
  return a;
}

// B-matrix 32x16 f16 layout: lanes 0-15 hold N=lane with K=k0..k0+15 packed
// 2-per-VGPR; lanes 16-31 hold same N with K=k0+16..k0+31. One contiguous
// 16-half (32B) load per lane from a (rows x stride) matrix where row = N
// and columns run along K (i.e. weight matrix W[o][c], or feature row f[j][c]
// for the Gram matrix since B = A^T there).
__device__ __forceinline__ v16h load_b_frag(const _Float16* base, int stride,
                                            int n0, int k0, int lane) {
  int n  = n0 + (lane & 15);
  int kk = k0 + ((lane & 16) ? 16 : 0);
  const _Float16* p = base + (size_t)n * stride + kk;
  v16h b;
#pragma unroll
  for (int i = 0; i < 16; ++i) b[i] = p[i];
  return b;
}

__device__ __forceinline__ v8f wmma_f16(v16h a, v16h b, v8f c) {
  // (neg_a, A, neg_b, B, c_mod, C, reuse_a, reuse_b)
  return __builtin_amdgcn_wmma_f32_16x16x32_f16(false, a, false, b, (short)0,
                                                c, false, false);
}

__device__ __forceinline__ v8f vzero8() {
  v8f z = {0.f, 0.f, 0.f, 0.f, 0.f, 0.f, 0.f, 0.f};
  return z;
}

// Order-preserving float -> uint encoding for LDS atomicMax on signed floats.
__device__ __forceinline__ unsigned enc_f32(float x) {
  unsigned b = __float_as_uint(x);
  return (b & 0x80000000u) ? ~b : (b | 0x80000000u);
}
__device__ __forceinline__ float dec_f32(unsigned e) {
  unsigned b = (e & 0x80000000u) ? (e & 0x7FFFFFFFu) : ~e;
  return __uint_as_float(b);
}

// ---------------- small prep kernels ----------------
// x (B,6,N) f32 -> xyzp / rgbp as (B,N,32) f16, channels 3..31 zero padded.
__global__ void prep_kernel(const float* __restrict__ x,
                            _Float16* __restrict__ xyzp,
                            _Float16* __restrict__ rgbp) {
  int i = blockIdx.x * blockDim.x + threadIdx.x;
  if (i >= BATCH * NPTS * 32) return;
  int c = i & 31, bn = i >> 5;
  int b = bn >> 12, n = bn & (NPTS - 1);
  float vx = 0.f, vr = 0.f;
  if (c < 3) {
    vx = x[((size_t)b * 6 + c) * NPTS + n];
    vr = x[((size_t)b * 6 + 3 + c) * NPTS + n];
  }
  xyzp[i] = (_Float16)vx;
  rgbp[i] = (_Float16)vr;
}

// f32 weights (Oreal,Cin) -> f16 (Opad,CinPad), zero padded.
__global__ void convw_kernel(const float* __restrict__ w,
                             _Float16* __restrict__ wh,
                             int Oreal, int Cin, int CinPad, int Opad) {
  int i = blockIdx.x * blockDim.x + threadIdx.x;
  if (i >= Opad * CinPad) return;
  int c = i % CinPad, o = i / CinPad;
  float v = (o < Oreal && c < Cin) ? w[(size_t)o * Cin + c] : 0.f;
  wh[i] = (_Float16)v;
}

// squared norms per point (rows of (B*N, stride) f16 matrix)
__global__ void sqnorm_kernel(const _Float16* __restrict__ feat, int stride,
                              int C, float* __restrict__ xx) {
  int bn = blockIdx.x * blockDim.x + threadIdx.x;
  if (bn >= BATCH * NPTS) return;
  const _Float16* p = feat + (size_t)bn * stride;
  float s = 0.f;
  for (int c = 0; c < C; ++c) { float f = (float)p[c]; s += f * f; }
  xx[bn] = s;
}

// ---------------- KNN: WMMA Gram tiles + register top-20 ----------------
// One wave per block; each wave owns 16 query rows, sweeps all 4096 columns
// in 16-wide tiles. Selection score s_j = 2<f_i,f_j> - ||f_j||^2 has the same
// per-row ordering as the reference's pd (the -||f_i||^2 row constant drops).
template <int KSTEPS>
__global__ void knn_kernel(const _Float16* __restrict__ feat, int fstride,
                           const float* __restrict__ xx, int* __restrict__ idx) {
  __shared__ float tile[16][17];
  int lane = threadIdx.x;
  int b = blockIdx.y;
  int row0 = blockIdx.x * 16;
  const _Float16* fb = feat + (size_t)b * NPTS * fstride;
  const float* xxb = xx + (size_t)b * NPTS;

  v16h afr[KSTEPS];
#pragma unroll
  for (int s = 0; s < KSTEPS; ++s)
    afr[s] = load_a_frag(fb, fstride, row0, s * 32, lane);

  float vals[KNN_K];
  int   inds[KNN_K];
#pragma unroll
  for (int t = 0; t < KNN_K; ++t) { vals[t] = -3.0e38f; inds[t] = 0; }

  for (int ct = 0; ct < NPTS / 16; ++ct) {
    v8f acc = vzero8();
#pragma unroll
    for (int s = 0; s < KSTEPS; ++s) {
      v16h bfr = load_b_frag(fb, fstride, ct * 16, s * 32, lane);
      acc = wmma_f16(afr[s], bfr, acc);
    }
    // C/D layout: lanes 0-15 VGPR r -> (M=r, N=lane); lanes 16-31 -> (M=8+r).
    int col = lane & 15;
    float xxj = xxb[ct * 16 + col];
    int roff = (lane & 16) ? 8 : 0;
#pragma unroll
    for (int r = 0; r < 8; ++r) tile[r + roff][col] = 2.f * acc[r] - xxj;
    __syncthreads();
    if (lane < 16) {  // each of 16 lanes owns one query row
#pragma unroll
      for (int j = 0; j < 16; ++j) {
        float s = tile[lane][j];
        if (s > vals[0]) {  // vals ascending; vals[0] = current worst kept
          vals[0] = s; inds[0] = ct * 16 + j;
#pragma unroll
          for (int t = 0; t < KNN_K - 1; ++t) {
            if (vals[t] > vals[t + 1]) {
              float tv = vals[t]; vals[t] = vals[t + 1]; vals[t + 1] = tv;
              int ti = inds[t]; inds[t] = inds[t + 1]; inds[t + 1] = ti;
            }
          }
        }
      }
    }
    __syncthreads();
  }
  if (lane < 16) {
    int* op = idx + ((size_t)b * NPTS + row0 + lane) * KNN_K;
#pragma unroll
    for (int t = 0; t < KNN_K; ++t) op[t] = inds[KNN_K - 1 - t];
  }
}

// ---------------- EdgeConv: gather e -> LDS, WMMA, BN+LReLU, max over K ----
// 8 points per block (160 edge rows), 256 threads = 8 waves. Max-over-K via
// order-preserving uint atomicMax in LDS.
__global__ void edgeconv_kernel(const _Float16* __restrict__ feat, int fstride,
                                int C, int Cpad, const int* __restrict__ idx,
                                const _Float16* __restrict__ wh, int O,
                                const float* __restrict__ gg,
                                const float* __restrict__ bb,
                                const float* __restrict__ mm,
                                const float* __restrict__ vv,
                                _Float16* __restrict__ out, int ostride) {
  __shared__ _Float16 e_lds[160 * 128];
  __shared__ unsigned maxbuf[8 * 128];
  int b = blockIdx.y;
  int p0 = blockIdx.x * 8;
  const _Float16* fb = feat + (size_t)b * NPTS * fstride;

  for (int i = threadIdx.x; i < 8 * O; i += 256) maxbuf[i] = 0u;  // enc(-inf)<... all encodings >= 0

  // stage e = [nbr - ctr (C), ctr (C), zero pad] rows: r = p*20 + k
  int total = 160 * Cpad;
  for (int i = threadIdx.x; i < total; i += 256) {
    int c = i % Cpad, r = i / Cpad;
    int p = r / KNN_K, k = r % KNN_K;
    int n = p0 + p;
    float val = 0.f;
    if (c < C) {
      int j = idx[((size_t)b * NPTS + n) * KNN_K + k];
      val = (float)fb[(size_t)j * fstride + c] - (float)fb[(size_t)n * fstride + c];
    } else if (c < 2 * C) {
      val = (float)fb[(size_t)n * fstride + (c - C)];
    }
    e_lds[r * Cpad + c] = (_Float16)val;
  }
  __syncthreads();

  int wave = threadIdx.x >> 5, lane = threadIdx.x & 31;
  int octiles = O >> 4;
  int ntiles = 10 * octiles;  // 160 rows / 16
  int ksteps = Cpad >> 5;
  for (int t = wave; t < ntiles; t += 8) {
    int mt = t % 10, ot = t / 10;
    v8f acc = vzero8();
    for (int s = 0; s < ksteps; ++s) {
      v16h a = load_a_frag(e_lds, Cpad, mt * 16, s * 32, lane);
      v16h w = load_b_frag(wh, Cpad, ot * 16, s * 32, lane);
      acc = wmma_f16(a, w, acc);
    }
    int o = ot * 16 + (lane & 15);
    float scale = gg[o] * rsqrtf(vv[o] + 1e-5f);
    float shift = bb[o] - mm[o] * scale;
    int roff = (lane & 16) ? 8 : 0;
#pragma unroll
    for (int r = 0; r < 8; ++r) {
      int row = mt * 16 + r + roff;
      int p = row / KNN_K;
      float h = acc[r] * scale + shift;
      h = (h >= 0.f) ? h : 0.2f * h;
      atomicMax(&maxbuf[p * O + o], enc_f32(h));
    }
  }
  __syncthreads();

  _Float16* ob = out + (size_t)b * NPTS * ostride;
  for (int i = threadIdx.x; i < 8 * O; i += 256) {
    int p = i / O, o = i % O;
    ob[(size_t)(p0 + p) * ostride + o] = (_Float16)dec_f32(maxbuf[i]);
  }
}

// ---------------- pointwise MLP GEMM (+BN+LReLU or bias) ----------------
// One wave per 16(point)x16(channel) output tile; grid (B*N/16, Opad/16).
__global__ void mlp_kernel(const _Float16* __restrict__ in, int istride,
                           int cinPad, const _Float16* __restrict__ wh,
                           const float* __restrict__ gg,
                           const float* __restrict__ bbn,
                           const float* __restrict__ mm,
                           const float* __restrict__ vv,
                           const float* __restrict__ bias,
                           _Float16* __restrict__ out16, int ostride,
                           float* __restrict__ out32, int o32mode, int Oreal) {
  int lane = threadIdx.x;
  int row0 = blockIdx.x * 16;
  int o0 = blockIdx.y * 16;
  v8f acc = vzero8();
  int ksteps = cinPad >> 5;
  for (int s = 0; s < ksteps; ++s) {
    v16h a = load_a_frag(in, istride, row0, s * 32, lane);
    v16h w = load_b_frag(wh, cinPad, o0, s * 32, lane);
    acc = wmma_f16(a, w, acc);
  }
  int o = o0 + (lane & 15);
  float scale, shift;
  bool doRelu;
  if (bias) {
    scale = 1.f; shift = (o < Oreal) ? bias[o] : 0.f; doRelu = false;
  } else {
    scale = gg[o] * rsqrtf(vv[o] + 1e-5f);
    shift = bbn[o] - mm[o] * scale; doRelu = true;
  }
  int roff = (lane & 16) ? 8 : 0;
#pragma unroll
  for (int r = 0; r < 8; ++r) {
    int row = row0 + r + roff;
    float h = acc[r] * scale + shift;
    if (doRelu) h = (h >= 0.f) ? h : 0.2f * h;
    if (out16) out16[(size_t)row * ostride + o] = (_Float16)h;
    if (out32) {
      int bb2 = row >> 12, n = row & (NPTS - 1);
      if (o32mode == 1)        // channel-major (B, Oreal, N)  -> x5
        out32[((size_t)bb2 * Oreal + o) * NPTS + n] = h;
      else if (o < Oreal)      // point-major (B, N, Oreal)    -> logits
        out32[(size_t)row * Oreal + o] = h;
    }
  }
}

// ---------------- host orchestration ----------------
extern "C" void kernel_launch(void* const* d_in, const int* in_sizes, int n_in,
                              void* d_out, int out_size, void* d_ws,
                              size_t ws_size, hipStream_t stream) {
  (void)in_sizes; (void)n_in; (void)out_size; (void)ws_size;
  const float* x = (const float*)d_in[0];
  // setup_inputs order: x, then {w,g,b,m,v} for ec1..ec4, col, c5..c7, then c8_w, c8_b
  const float* W[9]; const float* G[9]; const float* Bb[9];
  const float* M[9]; const float* V[9];
  for (int l = 0; l < 8; ++l) {
    W[l]  = (const float*)d_in[1 + 5 * l + 0];
    G[l]  = (const float*)d_in[1 + 5 * l + 1];
    Bb[l] = (const float*)d_in[1 + 5 * l + 2];
    M[l]  = (const float*)d_in[1 + 5 * l + 3];
    V[l]  = (const float*)d_in[1 + 5 * l + 4];
  }
  const float* c8_w = (const float*)d_in[41];
  const float* c8_b = (const float*)d_in[42];

  char* ws = (char*)d_ws;
  size_t off = 0;
  auto alloc = [&](size_t bytes) -> void* {
    void* p = ws + off;
    off += (bytes + 255) & ~(size_t)255;
    return p;
  };

  const int S = 1408;  // feature arena channel stride (=x_cat 384 + x5 1024)
  _Float16* F    = (_Float16*)alloc((size_t)BATCH * NPTS * S * 2);
  _Float16* xyzp = (_Float16*)alloc((size_t)BATCH * NPTS * 32 * 2);
  _Float16* rgbp = (_Float16*)alloc((size_t)BATCH * NPTS * 32 * 2);
  _Float16* x6b  = (_Float16*)alloc((size_t)BATCH * NPTS * 512 * 2);
  _Float16* x7b  = (_Float16*)alloc((size_t)BATCH * NPTS * 256 * 2);
  float*    xx   = (float*)alloc((size_t)BATCH * NPTS * 4);
  int*      idx  = (int*)alloc((size_t)BATCH * NPTS * KNN_K * 4);
  _Float16* wh[9];
  const int Wor[9]  = {64, 64, 64, 128, 64, 1024, 512, 256, 13};
  const int Wop[9]  = {64, 64, 64, 128, 64, 1024, 512, 256, 16};
  const int Wci[9]  = {6, 128, 128, 128, 3, 384, 1408, 512, 256};
  const int Wcp[9]  = {32, 128, 128, 128, 32, 384, 1408, 512, 256};
  for (int l = 0; l < 9; ++l)
    wh[l] = (_Float16*)alloc((size_t)Wop[l] * Wcp[l] * 2);

  // --- prep inputs + weights ---
  {
    int tot = BATCH * NPTS * 32;
    prep_kernel<<<(tot + 255) / 256, 256, 0, stream>>>(x, xyzp, rgbp);
  }
  for (int l = 0; l < 9; ++l) {
    const float* wsrc = (l < 8) ? W[l] : c8_w;
    int tot = Wop[l] * Wcp[l];
    convw_kernel<<<(tot + 255) / 256, 256, 0, stream>>>(wsrc, wh[l], Wor[l],
                                                        Wci[l], Wcp[l], Wop[l]);
  }

  // --- EdgeConv stack ---
  struct EC { const _Float16* feat; int fstride, C, Cpad, O, kst; _Float16* out; };
  EC ec[4] = {
    {xyzp,    32, 3,  32, 64,  1, F + 0},      // ec1 on xyz
    {F + 0,    S, 64, 128, 64, 2, F + 64},     // ec2 on x1
    {F + 64,   S, 64, 128, 64, 2, F + 128},    // ec3 on x2
    {F + 128,  S, 64, 128, 128, 2, F + 192},   // ec4 on x3
  };
  for (int l = 0; l < 4; ++l) {
    sqnorm_kernel<<<(BATCH * NPTS + 255) / 256, 256, 0, stream>>>(
        ec[l].feat, ec[l].fstride, ec[l].C, xx);
    dim3 kg(NPTS / 16, BATCH);
    if (ec[l].kst == 1)
      knn_kernel<1><<<kg, 32, 0, stream>>>(ec[l].feat, ec[l].fstride, xx, idx);
    else
      knn_kernel<2><<<kg, 32, 0, stream>>>(ec[l].feat, ec[l].fstride, xx, idx);
    dim3 eg(NPTS / 8, BATCH);
    edgeconv_kernel<<<eg, 256, 0, stream>>>(ec[l].feat, ec[l].fstride, ec[l].C,
                                            ec[l].Cpad, idx, wh[l], ec[l].O,
                                            G[l], Bb[l], M[l], V[l],
                                            ec[l].out, S);
  }

  // --- pointwise MLPs ---
  float* logits = (float*)d_out;                               // (B,N,13)
  float* x5out  = (float*)d_out + (size_t)BATCH * NPTS * 13;   // (B,1024,N)
  int rows16 = BATCH * NPTS / 16;

  // col: rgb(32pad) -> 64, into F channels 320..383
  mlp_kernel<<<dim3(rows16, 64 / 16), 32, 0, stream>>>(
      rgbp, 32, 32, wh[4], G[4], Bb[4], M[4], V[4], nullptr,
      F + 320, S, nullptr, 0, 64);
  // c5: x_cat(384) -> 1024, into F channels 384..1407 + f32 x5 (B,1024,N)
  mlp_kernel<<<dim3(rows16, 1024 / 16), 32, 0, stream>>>(
      F, S, 384, wh[5], G[5], Bb[5], M[5], V[5], nullptr,
      F + 384, S, x5out, 1, 1024);
  // c6: x_comb(1408) -> 512
  mlp_kernel<<<dim3(rows16, 512 / 16), 32, 0, stream>>>(
      F, S, 1408, wh[6], G[6], Bb[6], M[6], V[6], nullptr,
      x6b, 512, nullptr, 0, 512);
  // c7: 512 -> 256
  mlp_kernel<<<dim3(rows16, 256 / 16), 32, 0, stream>>>(
      x6b, 512, 512, wh[7], G[7], Bb[7], M[7], V[7], nullptr,
      x7b, 256, nullptr, 0, 256);
  // c8: 256 -> 13 (+bias only), f32 logits (B,N,13)
  mlp_kernel<<<dim3(rows16, 1), 32, 0, stream>>>(
      x7b, 256, 256, wh[8], nullptr, nullptr, nullptr, nullptr, c8_b,
      nullptr, 0, logits, 2, 13);
}
